// modi_cgcnn_angle_46248207843556
// MI455X (gfx1250) — matrix operand order repaired and checked by reading, not compiled, not run on hardware
//
#include <hip/hip_runtime.h>
#include <hip/hip_bf16.h>

#define EPSF 1e-5f
#define INV_SQRT2F 0.70710678118654752440f
#define NSEG 256

typedef __attribute__((ext_vector_type(16))) _Float16 v16h;
typedef __attribute__((ext_vector_type(8)))  _Float16 v8h;
typedef __attribute__((ext_vector_type(8)))  float    v8f;

__device__ __forceinline__ float siluf(float x) {
    return x * (1.0f / (1.0f + __expf(-x)));
}

// WMMA wrapper: D = A(16x32 f16) * B(32x16 f16) + C(16x16 f32)
__device__ __forceinline__ v8f wmma_f16(v16h a, v16h b, v8f c) {
    return __builtin_amdgcn_wmma_f32_16x16x32_f16(false, a, false, b, (short)0, c, false, false);
}

// A fragment (16x32, f16) from LDS, row-major with rowStride elements.
// Contiguous k order used here: lane<16 -> M=lane, K = k0+{0..15}
//             lane>=16 -> M=lane-16, K = k0+{16..31}
__device__ __forceinline__ v16h load_A_frag(const _Float16* sm, int lane, int rowStride, int k0) {
    int m  = lane & 15;
    int hs = (lane >> 4) * 16;
    const _Float16* p = sm + m * rowStride + k0 + hs;
    v8h a0 = *(const v8h*)(p);
    v8h a1 = *(const v8h*)(p + 8);
    return __builtin_shufflevector(a0, a1, 0,1,2,3,4,5,6,7,8,9,10,11,12,13,14,15);
}

// B fragment (32x16, f16) from global weights stored transposed: W[n][k], row length K.
// Contiguous k order, same as A: lane<16 -> col n0+lane, K = k0..k0+15 ; lane>=16 -> col n0+lane-16, K = k0+16..k0+31
__device__ __forceinline__ v16h load_B_frag(const _Float16* W, int lane, int K, int n0, int k0) {
    int n    = n0 + (lane & 15);
    int koff = k0 + (lane >> 4) * 16;
    return *(const v16h*)(W + (size_t)n * K + koff);
}

// ---------------- kernel 0: zero stats scratch ----------------
__global__ void zero_stats_kernel(float* p, int n) {
    int i = blockIdx.x * blockDim.x + threadIdx.x;
    if (i < n) p[i] = 0.0f;
}

// ---------------- kernel 1: weights -> f16, transposed [N][K] ----------------
__global__ void prep_weights_kernel(const float* w_full, const float* rw1, const float* rw2,
                                    _Float16* Wt, _Float16* w1t, _Float16* w2t) {
    int i = blockIdx.x * blockDim.x + threadIdx.x;
    if (i < 256 * 384) {                 // Wt[n(256)][k(384)] from w_full[k*256+n]
        int n = i / 384, k = i - n * 384;
        Wt[i] = (_Float16)w_full[k * 256 + n];
    }
    if (i < 2 * 64 * 128) {              // w1t[l][n(64)][k(128)] from rw1[l][k*64+n]
        int l = i / (64 * 128); int r = i - l * (64 * 128);
        int n = r / 128, k = r - n * 128;
        w1t[i] = (_Float16)rw1[l * 128 * 64 + k * 64 + n];
    }
    if (i < 2 * 128 * 64) {              // w2t[l][n(128)][k(64)] from rw2[l][k*128+n]
        int l = i / (128 * 64); int r = i - l * (128 * 64);
        int n = r / 64, k = r - n * 64;
        w2t[i] = (_Float16)rw2[l * 64 * 128 + k * 128 + n];
    }
}

// ---------------- kernel 2: gather + big GEMM: g = [angle | nb0 | nb1] @ W ----------------
// block = 128 threads (4 waves). Block handles 16 rows; wave w covers N in [w*64, w*64+64).
__global__ __launch_bounds__(128) void gemm1_kernel(const float* __restrict__ angle_fea,
                                                    const float* __restrict__ nbr_fea,
                                                    const int* __restrict__ nbr_idx,
                                                    const _Float16* __restrict__ Wt,
                                                    float* __restrict__ g) {
    __shared__ _Float16 As[16 * 384];
    const int row0 = blockIdx.x * 16;
    const int tid  = threadIdx.x;

    for (int e = tid; e < 16 * 384; e += 128) {
        int r = e / 384, c = e - r * 384;
        int row = row0 + r;
        float v;
        if (c < 128) {
            v = angle_fea[(size_t)row * 128 + c];
        } else if (c < 256) {
            int i0 = nbr_idx[row * 2 + 0];
            v = nbr_fea[(size_t)i0 * 128 + (c - 128)];
        } else {
            int i1 = nbr_idx[row * 2 + 1];
            v = nbr_fea[(size_t)i1 * 128 + (c - 256)];
        }
        As[e] = (_Float16)v;
    }
    __syncthreads();

    const int wave = tid >> 5;
    const int lane = tid & 31;
    const int nbase = wave * 64;

    v8f acc[4] = {};
    for (int kb = 0; kb < 384; kb += 32) {
        v16h a = load_A_frag(As, lane, 384, kb);
#pragma unroll
        for (int nt = 0; nt < 4; ++nt) {
            v16h b = load_B_frag(Wt, lane, 384, nbase + nt * 16, kb);
            acc[nt] = wmma_f16(a, b, acc[nt]);
        }
    }

    // D layout: VGPR r -> M = r + (lane>=16 ? 8 : 0); N = lane&15
    const int nlo = lane & 15;
    const int mb  = (lane >> 4) * 8;
#pragma unroll
    for (int nt = 0; nt < 4; ++nt) {
        int ncol = nbase + nt * 16 + nlo;
#pragma unroll
        for (int r = 0; r < 8; ++r) {
            int m = mb + r;
            g[(size_t)(row0 + m) * 256 + ncol] = acc[nt][r];
        }
    }
}

// ---------------- kernel 3: segment stats of g (sorted idx -> flush on change) ----------------
// blockDim.x == 256 (one thread per column). Each block scans R consecutive rows.
__global__ __launch_bounds__(256) void seg_stats_kernel(const float* __restrict__ x,
                                                        const int* __restrict__ idx,
                                                        float* __restrict__ ssum,
                                                        float* __restrict__ ssq,
                                                        float* __restrict__ cnt,
                                                        int n_rows, int R) {
    const int t  = threadIdx.x;
    const int r0 = blockIdx.x * R;
    int rend = r0 + R; if (rend > n_rows) rend = n_rows;
    if (r0 >= rend) return;

    int cur = idx[r0];
    int cstart = r0;
    float aS = 0.0f, aQ = 0.0f;
    for (int r = r0; r < rend; ++r) {
        int seg = idx[r];
        if (seg != cur) {
            atomicAdd(&ssum[cur * 256 + t], aS);
            atomicAdd(&ssq [cur * 256 + t], aQ);
            if (t == 0) atomicAdd(&cnt[cur], (float)(r - cstart));
            aS = 0.0f; aQ = 0.0f; cur = seg; cstart = r;
        }
        float v = x[(size_t)r * 256 + t];
        aS += v; aQ += v * v;
    }
    atomicAdd(&ssum[cur * 256 + t], aS);
    atomicAdd(&ssq [cur * 256 + t], aQ);
    if (t == 0) atomicAdd(&cnt[cur], (float)(rend - cstart));
}

// ---------------- kernel 4: norm1 + layernorm + silu + gate -> s ; stats of s ----------------
// blockDim.x == 128. Block processes R consecutive rows.
__global__ __launch_bounds__(128) void mid_kernel(const float* __restrict__ g,
                                                  const int* __restrict__ idx,
                                                  const float* __restrict__ sum1,
                                                  const float* __restrict__ sq1,
                                                  const float* __restrict__ cnt,
                                                  const float* __restrict__ n1w,
                                                  const float* __restrict__ n1b,
                                                  const float* __restrict__ lnw,
                                                  const float* __restrict__ lnb,
                                                  const float* __restrict__ maskw,
                                                  float* __restrict__ s,
                                                  float* __restrict__ sum2,
                                                  float* __restrict__ sq2,
                                                  int n_rows, int R) {
    __shared__ float redA[128];
    __shared__ float redB[128];
    const int t  = threadIdx.x;
    const int r0 = blockIdx.x * R;
    int rend = r0 + R; if (rend > n_rows) rend = n_rows;
    if (r0 >= rend) return;

    const float mw = maskw[t];
    const float w0 = n1w[t],       b0 = n1b[t];
    const float w1 = n1w[128 + t], b1 = n1b[128 + t];
    const float lw = lnw[t],       lb = lnb[t];

    int cur = idx[r0];
    float aS = 0.0f, aQ = 0.0f;

    for (int r = r0; r < rend; ++r) {
        int seg = idx[r];
        if (seg != cur) {
            atomicAdd(&sum2[cur * 128 + t], aS);
            atomicAdd(&sq2 [cur * 128 + t], aQ);
            aS = 0.0f; aQ = 0.0f; cur = seg;
        }
        float invc = 1.0f / fmaxf(cnt[seg], 1.0f);
        float m0 = sum1[seg * 256 + t] * invc;
        float v0 = sq1 [seg * 256 + t] * invc - m0 * m0;
        float m1 = sum1[seg * 256 + 128 + t] * invc;
        float v1 = sq1 [seg * 256 + 128 + t] * invc - m1 * m1;

        float gc = g[(size_t)r * 256 + t];
        float gf = g[(size_t)r * 256 + 128 + t];
        float core = (gc - m0) * rsqrtf(v0 + EPSF) * w0 + b0;
        float filt = (gf - m1) * rsqrtf(v1 + EPSF) * w1 + b1;

        // layer_norm over the 128 core values (one per thread)
        redA[t] = core; redB[t] = core * core;
        __syncthreads();
        for (int off = 64; off > 0; off >>= 1) {
            if (t < off) { redA[t] += redA[t + off]; redB[t] += redB[t + off]; }
            __syncthreads();
        }
        float S = redA[0], Q = redB[0];
        __syncthreads();
        float mu = S * (1.0f / 128.0f);
        float vv = Q * (1.0f / 128.0f) - mu * mu;
        float cn = (core - mu) * rsqrtf(vv + EPSF) * lw + lb;
        float ca = siluf(cn);

        // gate = sigmoid(dot(filt, mask_w))
        redA[t] = filt * mw;
        __syncthreads();
        for (int off = 64; off > 0; off >>= 1) {
            if (t < off) redA[t] += redA[t + off];
            __syncthreads();
        }
        float gsum = redA[0];
        __syncthreads();
        float gate = 1.0f / (1.0f + __expf(-gsum));

        float sval = gate * ca;
        s[(size_t)r * 128 + t] = sval;
        aS += sval; aQ += sval * sval;
    }
    atomicAdd(&sum2[cur * 128 + t], aS);
    atomicAdd(&sq2 [cur * 128 + t], aQ);
}

// ---------------- kernel 5: norm2 + 2x residual MLP (WMMA) + output ----------------
// block = 1 wave (32 threads), handles 16 rows.
__global__ __launch_bounds__(32) void tail_kernel(const float* __restrict__ s,
                                                  const int* __restrict__ idx,
                                                  const float* __restrict__ sum2,
                                                  const float* __restrict__ sq2,
                                                  const float* __restrict__ cnt,
                                                  const float* __restrict__ n2w,
                                                  const float* __restrict__ n2b,
                                                  const _Float16* __restrict__ w1t,
                                                  const float* __restrict__ rb1,
                                                  const _Float16* __restrict__ w2t,
                                                  const float* __restrict__ rb2,
                                                  const float* __restrict__ angle_fea,
                                                  float* __restrict__ out) {
    __shared__ float    sN[16 * 128];
    __shared__ _Float16 sH[16 * 128];
    __shared__ _Float16 hH[16 * 64];
    const int lane = threadIdx.x;
    const int row0 = blockIdx.x * 16;

    for (int e = lane; e < 16 * 128; e += 32) {
        int r = e >> 7, c = e & 127;
        int row = row0 + r;
        int seg = idx[row];
        float invc = 1.0f / fmaxf(cnt[seg], 1.0f);
        float m = sum2[seg * 128 + c] * invc;
        float v = sq2 [seg * 128 + c] * invc - m * m;
        float val = (s[(size_t)row * 128 + c] - m) * rsqrtf(v + EPSF) * n2w[c] + n2b[c];
        sN[e] = val;
        sH[e] = (_Float16)val;
    }
    __syncthreads();

    const int nlo = lane & 15;
    const int mb  = (lane >> 4) * 8;

    for (int l = 0; l < 2; ++l) {
        const _Float16* W1 = w1t + l * 64 * 128;
        const _Float16* W2 = w2t + l * 128 * 64;

        // h = silu(sN @ W1 + b1)   (16x128 @ 128x64)
        v8f acc1[4] = {};
        for (int kb = 0; kb < 128; kb += 32) {
            v16h a = load_A_frag(sH, lane, 128, kb);
#pragma unroll
            for (int nt = 0; nt < 4; ++nt) {
                v16h b = load_B_frag(W1, lane, 128, nt * 16, kb);
                acc1[nt] = wmma_f16(a, b, acc1[nt]);
            }
        }
#pragma unroll
        for (int nt = 0; nt < 4; ++nt) {
            int n = nt * 16 + nlo;
            float bias = rb1[l * 64 + n];
#pragma unroll
            for (int r8 = 0; r8 < 8; ++r8) {
                int m = mb + r8;
                float x = acc1[nt][r8] + bias;
                hH[m * 64 + n] = (_Float16)siluf(x);
            }
        }
        __syncthreads();

        // s = (s + silu(h @ W2 + b2)) * INV_SQRT2   (16x64 @ 64x128)
        v8f acc2[8] = {};
        for (int kb = 0; kb < 64; kb += 32) {
            v16h a = load_A_frag(hH, lane, 64, kb);
#pragma unroll
            for (int nt = 0; nt < 8; ++nt) {
                v16h b = load_B_frag(W2, lane, 64, nt * 16, kb);
                acc2[nt] = wmma_f16(a, b, acc2[nt]);
            }
        }
#pragma unroll
        for (int nt = 0; nt < 8; ++nt) {
            int n = nt * 16 + nlo;
            float bias = rb2[l * 128 + n];
#pragma unroll
            for (int r8 = 0; r8 < 8; ++r8) {
                int m = mb + r8;
                float x = acc2[nt][r8] + bias;
                float sn = (sN[m * 128 + n] + siluf(x)) * INV_SQRT2F;
                sN[m * 128 + n] = sn;
                sH[m * 128 + n] = (_Float16)sn;
            }
        }
        __syncthreads();
    }

    for (int e = lane; e < 16 * 128; e += 32) {
        int r = e >> 7, c = e & 127;
        int row = row0 + r;
        out[(size_t)row * 128 + c] = INV_SQRT2F * (angle_fea[(size_t)row * 128 + c] + sN[e]);
    }
}

extern "C" void kernel_launch(void* const* d_in, const int* in_sizes, int n_in,
                              void* d_out, int out_size, void* d_ws, size_t ws_size,
                              hipStream_t stream) {
    const float* nbr_fea   = (const float*)d_in[0];
    const float* angle_fea = (const float*)d_in[1];
    const int*   nbr_idx   = (const int*)d_in[2];
    const int*   cidx      = (const int*)d_in[3];
    const float* w_full    = (const float*)d_in[4];
    const float* n1w       = (const float*)d_in[5];
    const float* n1b       = (const float*)d_in[6];
    const float* lnw       = (const float*)d_in[7];
    const float* lnb       = (const float*)d_in[8];
    const float* maskw     = (const float*)d_in[9];
    const float* n2w       = (const float*)d_in[10];
    const float* n2b       = (const float*)d_in[11];
    const float* rw1       = (const float*)d_in[12];
    const float* rb1       = (const float*)d_in[13];
    const float* rw2       = (const float*)d_in[14];
    const float* rb2       = (const float*)d_in[15];

    const int NA = in_sizes[3];            // number of angle rows (1,000,000)

    // workspace layout (floats)
    float* g    = (float*)d_ws;                        // NA*256
    float* sbuf = g    + (size_t)NA * 256;             // NA*128
    float* sum1 = sbuf + (size_t)NA * 128;             // 256*256
    float* sq1  = sum1 + NSEG * 256;                   // 256*256
    float* sum2 = sq1  + NSEG * 256;                   // 256*128
    float* sq2  = sum2 + NSEG * 128;                   // 256*128
    float* cnt  = sq2  + NSEG * 128;                   // 256
    _Float16* Wt  = (_Float16*)(cnt + NSEG);           // 256*384
    _Float16* w1t = Wt  + 256 * 384;                   // 2*64*128
    _Float16* w2t = w1t + 2 * 64 * 128;                // 2*128*64

    // 0) zero stats accumulators (sum1,sq1,sum2,sq2,cnt are contiguous)
    int nz = NSEG * 256 * 2 + NSEG * 128 * 2 + NSEG;
    zero_stats_kernel<<<(nz + 255) / 256, 256, 0, stream>>>(sum1, nz);

    // 1) weight transpose + f16 convert
    prep_weights_kernel<<<(256 * 384 + 255) / 256, 256, 0, stream>>>(w_full, rw1, rw2, Wt, w1t, w2t);

    // 2) gather + big WMMA GEMM -> g
    gemm1_kernel<<<NA / 16, 128, 0, stream>>>(angle_fea, nbr_fea, nbr_idx, Wt, g);

    // 3) segment stats of g (+counts)
    {
        int R = 128;
        seg_stats_kernel<<<(NA + R - 1) / R, 256, 0, stream>>>(g, cidx, sum1, sq1, cnt, NA, R);
    }

    // 4) norm1 + layernorm + silu + gate -> s, plus stats of s
    {
        int R = 64;
        mid_kernel<<<(NA + R - 1) / R, 128, 0, stream>>>(g, cidx, sum1, sq1, cnt,
                                                         n1w, n1b, lnw, lnb, maskw,
                                                         sbuf, sum2, sq2, NA, R);
    }

    // 5) norm2 + residual WMMA MLPs + output
    tail_kernel<<<NA / 16, 32, 0, stream>>>(sbuf, cidx, sum2, sq2, cnt, n2w, n2b,
                                            w1t, rb1, w2t, rb2, angle_fea, (float*)d_out);
}
